// CausalMultiHeadAttention_82325933129936
// MI455X (gfx1250) — compile-verified, matrix operand order checked by reading, not run on hardware
//
#include <hip/hip_runtime.h>
#include <hip/hip_bf16.h>

typedef __attribute__((ext_vector_type(16))) __bf16 v16bf;
typedef __attribute__((ext_vector_type(8)))  __bf16 v8bf;
typedef __attribute__((ext_vector_type(8)))  float  v8f;

#define DEV __device__ __forceinline__

static constexpr int Bb = 4;
static constexpr int Ss = 2048;
static constexpr int Dd = 1024;
static constexpr int Hh = 16;
static constexpr int DK = 64;
static constexpr int SV = Ss + 32;   // padded key stride for V^T rows

DEV v8bf ld8(const __bf16* p) { return *reinterpret_cast<const v8bf*>(p); }

DEV v16bf mk16(v8bf lo, v8bf hi) {
  return __builtin_shufflevector(lo, hi, 0,1,2,3,4,5,6,7,8,9,10,11,12,13,14,15);
}

DEV v8f wmma_bf16(v16bf a, v16bf b, v8f c) {
  return __builtin_amdgcn_wmma_f32_16x16x32_bf16(false, a, false, b, (short)0, c,
                                                 false, false);
}

// Row reductions across the 16-lane half-groups of the C-tile layout.
DEV float rmax16(float v) {
  v = fmaxf(v, __shfl_xor(v, 1, 32));
  v = fmaxf(v, __shfl_xor(v, 2, 32));
  v = fmaxf(v, __shfl_xor(v, 4, 32));
  v = fmaxf(v, __shfl_xor(v, 8, 32));
  return v;
}
DEV float rsum16(float v) {
  v += __shfl_xor(v, 1, 32);
  v += __shfl_xor(v, 2, 32);
  v += __shfl_xor(v, 4, 32);
  v += __shfl_xor(v, 8, 32);
  return v;
}

__global__ void cvt_bf16_kernel(const float* __restrict__ x, __bf16* __restrict__ y, int n) {
  int i = blockIdx.x * blockDim.x + threadIdx.x;
  int st = gridDim.x * blockDim.x;
  for (; i < n; i += st) y[i] = (__bf16)x[i];
}

__global__ void zero_bf16_kernel(__bf16* __restrict__ y, int n) {
  int i = blockIdx.x * blockDim.x + threadIdx.x;
  int st = gridDim.x * blockDim.x;
  for (; i < n; i += st) y[i] = (__bf16)0.0f;
}

// C = A[M x 1024] * W^T (W stored [1024 x 1024] row-major, torch Linear semantics).
// One wave computes a 64(M) x 64(N) tile: 16 WMMA per 16 b128 loads per K-step.
// mode 0: Q/K projection -> RoPE -> bf16 [B,H,S,64]
// mode 1: V projection   -> bf16 V^T [B,H,64,SV]
// mode 2: O projection   -> f32 [B,S,D]
__global__ __launch_bounds__(32)
void gemm64x64_kernel(const __bf16* __restrict__ A, const __bf16* __restrict__ W,
                      void* __restrict__ out, const int* __restrict__ tokpos, int mode) {
  const int Kd = Dd;
  const unsigned lane  = threadIdx.x & 31u;
  const unsigned ln15  = lane & 15u;
  const int rowoff     = (int)((lane >> 4) << 3);
  const int koff       = (lane & 16u) ? 8 : 0;
  const int NT = Dd / 64;
  const int mt = blockIdx.x / NT;
  const int nt = blockIdx.x - mt * NT;
  const int m0 = mt * 64, n0 = nt * 64;

  const __bf16* ap[4];
  const __bf16* wp[4];
#pragma unroll
  for (int i = 0; i < 4; ++i) {
    ap[i] = A + (size_t)(m0 + i * 16 + (int)ln15) * Kd;
    wp[i] = W + (size_t)(n0 + i * 16 + (int)ln15) * Kd;
  }

  v8f zero = {};
  v8f acc[4][4];
#pragma unroll
  for (int i = 0; i < 4; ++i)
#pragma unroll
    for (int t = 0; t < 4; ++t) acc[i][t] = zero;

  for (int k = 0; k < Kd; k += 32) {
    const int ko = k + koff;
    v16bf a[4];
#pragma unroll
    for (int i = 0; i < 4; ++i)
      a[i] = mk16(ld8(ap[i] + ko), ld8(ap[i] + ko + 16));
#pragma unroll
    for (int t = 0; t < 4; ++t) {
      v16bf b = mk16(ld8(wp[t] + ko), ld8(wp[t] + ko + 16));
#pragma unroll
      for (int i = 0; i < 4; ++i)
        acc[i][t] = wmma_bf16(a[i], b, acc[i][t]);
    }
  }

  const int b = m0 >> 11;  // row / S (tile never crosses batch: 64 | 2048)
#pragma unroll
  for (int hl = 0; hl < 4; ++hl) {
    const int mb = m0 + hl * 16;
#pragma unroll
    for (int t = 0; t < 4; ++t) {
      const int n = n0 + t * 16 + (int)ln15;
      if (mode == 2) {
        float* of = (float*)out;
#pragma unroll
        for (int r = 0; r < 8; ++r)
          of[(size_t)(mb + r + rowoff) * Dd + n] = acc[hl][t][r];
      } else {
        __bf16* ob = (__bf16*)out;
        const int h  = n >> 6;
        const int dh = n & 63;
#pragma unroll
        for (int r = 0; r < 8; ++r) {
          const int m   = mb + r + rowoff;
          const int pos = m & (Ss - 1);
          float v = acc[hl][t][r];
          if (mode == 0) {
            // interleaved-pair RoPE: partner dim lives in adjacent lane
            float pv = __shfl_xor(v, 1, 32);
            const int pi = dh >> 1;
            float freq = __expf(-0.28782313662f * (float)pi);  // ln(1e4)/32
            float ang  = (float)tokpos[pos] * freq;
            float sn, cs;
            __sincosf(ang, &sn, &cs);
            v = (dh & 1) ? fmaf(pv, sn, v * cs) : fmaf(v, cs, -pv * sn);
            ob[((size_t)(b * Hh + h) * Ss + pos) * DK + dh] = (__bf16)v;
          } else {  // mode 1: V^T store, contiguous along S per lane
            ob[((size_t)(b * Hh + h) * DK + dh) * SV + pos] = (__bf16)v;
          }
        }
      }
    }
  }
}

// Flash attention: 1 wave per 32-row query tile (two 16-row sub-tiles share all
// K/V fragments), 32 keys per step -> 16 WMMA per 16 b128 loads.
__global__ __launch_bounds__(32)
void attn_kernel(const __bf16* __restrict__ Q, const __bf16* __restrict__ K,
                 const __bf16* __restrict__ Vt, __bf16* __restrict__ O) {
  const unsigned lane = threadIdx.x & 31u;
  const unsigned ln15 = lane & 15u;
  const int rowoff    = (int)((lane >> 4) << 3);
  const int koff      = (lane & 16u) ? 8 : 0;

  const int qtiles = Ss >> 5;           // 32-row tiles
  const int bh = blockIdx.x / qtiles;
  const int qt = blockIdx.x - bh * qtiles;
  const int q0 = qt << 5;

  const __bf16* Qbh = Q  + (size_t)bh * Ss * DK;
  const __bf16* Kbh = K  + (size_t)bh * Ss * DK;
  const __bf16* Vbh = Vt + (size_t)bh * DK * SV;

  v16bf aq[2][2];
#pragma unroll
  for (int qh = 0; qh < 2; ++qh) {
    const __bf16* qr = Qbh + (size_t)(q0 + qh * 16 + (int)ln15) * DK;
    aq[qh][0] = mk16(ld8(qr + koff),      ld8(qr + koff + 16));
    aq[qh][1] = mk16(ld8(qr + 32 + koff), ld8(qr + 32 + koff + 16));
  }

  v8f zero = {};
  v8f acc[2][4];
  float mi[2][8], li[2][8];
#pragma unroll
  for (int qh = 0; qh < 2; ++qh) {
#pragma unroll
    for (int t = 0; t < 4; ++t) acc[qh][t] = zero;
#pragma unroll
    for (int r = 0; r < 8; ++r) { mi[qh][r] = -__builtin_inff(); li[qh][r] = 0.0f; }
  }

  __shared__ alignas(16) __bf16 Plds[2 * 16 * 32];

  const float sc = 0.125f;  // 1/sqrt(64)

  for (int k0 = 0; k0 <= q0 + 31; k0 += 32) {
    // --- scores: 2 key-tiles x 2 query-sub-tiles (K frags shared) ---
    v8f s[2][2];
#pragma unroll
    for (int kt = 0; kt < 2; ++kt) {
      int kr = k0 + kt * 16 + (int)ln15;
      kr = (kr > Ss - 1) ? (Ss - 1) : kr;   // clamp (masked anyway)
      const __bf16* kp = Kbh + (size_t)kr * DK;
      v16bf b0 = mk16(ld8(kp + koff),      ld8(kp + koff + 16));
      v16bf b1 = mk16(ld8(kp + 32 + koff), ld8(kp + 32 + koff + 16));
#pragma unroll
      for (int qh = 0; qh < 2; ++qh) {
        v8f c = zero;
        c = wmma_bf16(aq[qh][0], b0, c);
        c = wmma_bf16(aq[qh][1], b1, c);
        s[qh][kt] = c;
      }
    }
    // prefetch next K rows into cache
    {
      int nk = k0 + 32 + (int)ln15;
      nk = (nk > Ss - 1) ? (Ss - 1) : nk;
      __builtin_prefetch(Kbh + (size_t)nk * DK, 0, 1);
    }

    const int key0 = k0 + (int)ln15;
    const int key1 = k0 + 16 + (int)ln15;
#pragma unroll
    for (int qh = 0; qh < 2; ++qh) {
#pragma unroll
      for (int r = 0; r < 8; ++r) {
        const int qrow = q0 + qh * 16 + r + rowoff;
        float x0 = (key0 > qrow) ? -__builtin_inff() : s[qh][0][r] * sc;
        float x1 = (key1 > qrow) ? -__builtin_inff() : s[qh][1][r] * sc;
        float mt = rmax16(fmaxf(x0, x1));
        float mn = fmaxf(mi[qh][r], mt);
        float al = __expf(mi[qh][r] - mn);
        mi[qh][r] = mn;
        float p0 = __expf(x0 - mn);
        float p1 = __expf(x1 - mn);
        li[qh][r] = li[qh][r] * al + rsum16(p0 + p1);
#pragma unroll
        for (int t = 0; t < 4; ++t) acc[qh][t][r] *= al;
        // stage P (f32 C-layout -> bf16 A-layout) through LDS
        const int row = r + rowoff;
        Plds[qh * 512 + row * 32 + (int)ln15]      = (__bf16)p0;
        Plds[qh * 512 + row * 32 + 16 + (int)ln15] = (__bf16)p1;
      }
    }
    __syncthreads();
    v16bf ap0 = mk16(ld8(&Plds[(int)ln15 * 32 + koff]),
                     ld8(&Plds[(int)ln15 * 32 + koff + 16]));
    v16bf ap1 = mk16(ld8(&Plds[512 + (int)ln15 * 32 + koff]),
                     ld8(&Plds[512 + (int)ln15 * 32 + koff + 16]));

    // --- O += P @ V : V frags shared by both query sub-tiles ---
#pragma unroll
    for (int t = 0; t < 4; ++t) {
      const __bf16* vp = Vbh + (size_t)(t * 16 + (int)ln15) * SV + k0;
      v16bf bv = mk16(ld8(vp + koff), ld8(vp + koff + 16));
      acc[0][t] = wmma_bf16(ap0, bv, acc[0][t]);
      acc[1][t] = wmma_bf16(ap1, bv, acc[1][t]);
    }
  }

  const int bq = bh / Hh;
  const int hq = bh - bq * Hh;
#pragma unroll
  for (int qh = 0; qh < 2; ++qh) {
#pragma unroll
    for (int r = 0; r < 8; ++r) {
      const float inv = 1.0f / li[qh][r];
      const int qrow = q0 + qh * 16 + r + rowoff;
#pragma unroll
      for (int t = 0; t < 4; ++t) {
        O[((size_t)bq * Ss + qrow) * Dd + hq * DK + t * 16 + (int)ln15] =
            (__bf16)(acc[qh][t][r] * inv);
      }
    }
  }
}

extern "C" void kernel_launch(void* const* d_in, const int* in_sizes, int n_in,
                              void* d_out, int out_size, void* d_ws, size_t ws_size,
                              hipStream_t stream) {
  (void)in_sizes; (void)n_in; (void)out_size; (void)ws_size;
  const float* x  = (const float*)d_in[0];
  const float* Wq = (const float*)d_in[1];
  const float* Wk = (const float*)d_in[2];
  const float* Wv = (const float*)d_in[3];
  const float* Wo = (const float*)d_in[4];
  const int* tokpos = (const int*)d_in[6];

  size_t off = 0;
  auto carve = [&](size_t bytes) -> void* {
    void* p = (char*)d_ws + off;
    off += (bytes + 255) & ~(size_t)255;
    return p;
  };
  const size_t MR = (size_t)Bb * Ss;  // 8192 rows
  __bf16* xb  = (__bf16*)carve(MR * Dd * 2);          // reused as attnOut later
  __bf16* wqb = (__bf16*)carve((size_t)Dd * Dd * 2);
  __bf16* wkb = (__bf16*)carve((size_t)Dd * Dd * 2);
  __bf16* wvb = (__bf16*)carve((size_t)Dd * Dd * 2);
  __bf16* wob = (__bf16*)carve((size_t)Dd * Dd * 2);
  __bf16* Qb  = (__bf16*)carve(MR * Dd * 2);          // [B,H,S,64]
  __bf16* Kb  = (__bf16*)carve(MR * Dd * 2);          // [B,H,S,64]
  __bf16* Vt  = (__bf16*)carve((size_t)Bb * Hh * DK * SV * 2);  // [B,H,64,SV]

  const int nX = (int)(MR * Dd);
  const int nW = Dd * Dd;
  cvt_bf16_kernel<<<512, 256, 0, stream>>>(x,  xb,  nX);
  cvt_bf16_kernel<<<64,  256, 0, stream>>>(Wq, wqb, nW);
  cvt_bf16_kernel<<<64,  256, 0, stream>>>(Wk, wkb, nW);
  cvt_bf16_kernel<<<64,  256, 0, stream>>>(Wv, wvb, nW);
  cvt_bf16_kernel<<<64,  256, 0, stream>>>(Wo, wob, nW);
  zero_bf16_kernel<<<512, 256, 0, stream>>>(Vt, Bb * Hh * DK * SV);

  const int gproj = (int)(MR / 64) * (Dd / 64);  // 2048 waves
  gemm64x64_kernel<<<gproj, 32, 0, stream>>>(xb, wqb, Qb, tokpos, 0);
  gemm64x64_kernel<<<gproj, 32, 0, stream>>>(xb, wkb, Kb, tokpos, 0);
  gemm64x64_kernel<<<gproj, 32, 0, stream>>>(xb, wvb, Vt, tokpos, 1);

  __bf16* attnO = xb;  // xb dead after projections
  const int gattn = Bb * Hh * (Ss / 32);  // 4096 waves
  attn_kernel<<<gattn, 32, 0, stream>>>(Qb, Kb, Vt, attnO);

  gemm64x64_kernel<<<gproj, 32, 0, stream>>>(attnO, wob, d_out, tokpos, 2);
}
